// MPNEncoder_58394375356579
// MI455X (gfx1250) — matrix operand order repaired and checked by reading, not compiled, not run on hardware
//
#include <hip/hip_runtime.h>
#include <hip/hip_bf16.h>

// MPNEncoder for MI455X (gfx1250): wave32 WMMA f32<-f16 GEMMs with fused
// message construction / relu / softmax epilogues. Bandwidth-bound design:
// minimize passes over the 157MB message_bond tensors. Column tiles are
// compile-time (template NCT) so the WMMA inner loop is branch-free.

typedef __attribute__((ext_vector_type(16))) _Float16 v16h;
typedef __attribute__((ext_vector_type(8)))  float    v8f;

#define CTILES 19      // ceil(300/16)
#define HH 300

__device__ __forceinline__ v8f wmma_f16(v16h a, v16h b, v8f c) {
  // D = A(16x32 f16) * B(32x16 f16) + C(16x16 f32)
  return __builtin_amdgcn_wmma_f32_16x16x32_f16(false, a, false, b, (short)0, c,
                                                false, false);
}
__device__ __forceinline__ float relu_f(float x) { return x > 0.f ? x : 0.f; }
__device__ __forceinline__ float sigmoid_f(float x) { return 1.f / (1.f + __expf(-x)); }

// ---------------------------------------------------------------------------
// Pack a row-major f32 weight [K, 300] into per-(ktile, coltile, lane) WMMA
// B fragments (v16h = 32B each). B-frag layout (16-bit B, 32x16):
//   lane 0-15  : col = lane,     elems 0..7 = K (kt*32 + 0..7),  8..15 = K (kt*32 + 16..23)
//   lane 16-31 : col = lane-16,  elems 0..7 = K (kt*32 + 8..15), 8..15 = K (kt*32 + 24..31)
// ---------------------------------------------------------------------------
__global__ void k_pack_w(const float* __restrict__ W, _Float16* __restrict__ P,
                         int K, int KT) {
  int idx = blockIdx.x * blockDim.x + threadIdx.x;
  int total = KT * CTILES * 32;
  if (idx >= total) return;
  int lane = idx & 31;
  int t = idx >> 5;
  int ct = t % CTILES;
  int kt = t / CTILES;
  int col = ct * 16 + (lane & 15);
  int kb = kt * 32 + ((lane >> 4) * 8);
  v16h v;
#pragma unroll
  for (int i = 0; i < 16; ++i) {
    int k = kb + i + (i >= 8 ? 8 : 0);
    float x = (k < K && col < HH) ? W[k * HH + col] : 0.f;
    v[i] = (_Float16)x;
  }
  ((v16h*)P)[idx] = v;
}

// ---------------------------------------------------------------------------
// Generic input GEMM: out = relu( (scale[r] * A[r,:]) @ W ), A row-major f32.
// A-frag layout (16-bit A, 16x32): lane holds row (lane&15);
//   kOff = (lane>>4)*8; elems 0..7 = K(k0+kOff+0..7), elems 8..15 = +16.
// One wave per 16-row tile; NCT column tiles per wave starting at ctBase.
// KTfull tiles need no K-guard; tiles [KTfull, KT) are element-guarded.
// ---------------------------------------------------------------------------
template <int NCT>
__global__ void k_gemm_in(const float* __restrict__ A,
                          const float* __restrict__ rowScale,
                          const _Float16* __restrict__ Wp,
                          float* __restrict__ out, int ctBase, int K, int lda,
                          int KTfull, int KT) {
  int rt = blockIdx.x;
  int lane = threadIdx.x;
  int kOff = (lane >> 4) * 8;
  int r = rt * 16 + (lane & 15);
  const float* Arow = A + (size_t)r * lda;
  float sc = rowScale ? rowScale[r] : 1.f;
  v8f zero = {};
  v8f acc[NCT];
#pragma unroll
  for (int c = 0; c < NCT; ++c) acc[c] = zero;
  const v16h* Bbase = (const v16h*)Wp + lane + (size_t)ctBase * 32;

  for (int kt = 0; kt < KTfull; ++kt) {
    int k0 = kt * 32 + kOff;
    v16h a;
#pragma unroll
    for (int g = 0; g < 2; ++g)
#pragma unroll
      for (int t = 0; t < 8; ++t)
        a[g * 8 + t] = (_Float16)(sc * Arow[k0 + g * 16 + t]);
    const v16h* B = Bbase + (size_t)kt * (CTILES * 32);
    v16h bf[NCT];
#pragma unroll
    for (int c = 0; c < NCT; ++c) bf[c] = B[c * 32];
#pragma unroll
    for (int c = 0; c < NCT; ++c) acc[c] = wmma_f16(a, bf[c], acc[c]);
  }
  for (int kt = KTfull; kt < KT; ++kt) {  // guarded tail tile(s)
    int k0 = kt * 32 + kOff;
    v16h a;
#pragma unroll
    for (int g = 0; g < 2; ++g)
#pragma unroll
      for (int t = 0; t < 8; ++t) {
        int k = k0 + g * 16 + t;
        a[g * 8 + t] = (_Float16)((k < K) ? sc * Arow[k] : 0.f);
      }
    const v16h* B = Bbase + (size_t)kt * (CTILES * 32);
    v16h bf[NCT];
#pragma unroll
    for (int c = 0; c < NCT; ++c) bf[c] = B[c * 32];
#pragma unroll
    for (int c = 0; c < NCT; ++c) acc[c] = wmma_f16(a, bf[c], acc[c]);
  }

  // C/D layout: vgpr v, lanes0-15 -> row v, col lane; lanes16-31 -> row v+8
  int vrow = (lane >> 4) * 8;
#pragma unroll
  for (int c = 0; c < NCT; ++c) {
    int col = (ctBase + c) * 16 + (lane & 15);
    if (col >= HH) continue;
#pragma unroll
    for (int v = 0; v < 8; ++v) {
      int rr = rt * 16 + vrow + v;
      out[(size_t)rr * HH + col] = relu_f(acc[c][v]);
    }
  }
}

// ---------------------------------------------------------------------------
// Fused edge-update GEMM (the two big K=300 GEMMs):
//   A(r,k) with r=((bs*64+i)*64+j) is built in registers:
//       adj[bs,j,i]*msg_atom[bs,i,k] - msg_in[bs,j,i,k]   (transposed mb)
//   out[r,c] = relu(input_bond[r,c] + (A@Wh)[r,c]) * softmax[bs,i,j]
// A rows are 16B aligned (stride 300 f32) -> float4 loads for full tiles.
// ---------------------------------------------------------------------------
template <int NCT>
__global__ void k_edge_gemm(const float* __restrict__ msgIn,
                            const float* __restrict__ msgAtom,
                            const float* __restrict__ adj,
                            const float* __restrict__ inputBond,
                            const float* __restrict__ sm,
                            const _Float16* __restrict__ Wp,
                            float* __restrict__ msgOut, int ctBase) {
  int rt = blockIdx.x;           // 0..8191 ; rt*16 = bs*4096 + i*64 + j0
  int lane = threadIdx.x;
  int bs = rt >> 8;
  int i = (rt >> 2) & 63;
  int j0 = (rt & 3) * 16;
  int kOff = (lane >> 4) * 8;
  int j = j0 + (lane & 15);

  float adjv = adj[(bs * 64 + j) * 64 + i];
  const float* maRow = msgAtom + (size_t)(bs * 64 + i) * HH;
  const float* mbRow = msgIn + (size_t)((bs * 64 + j) * 64 + i) * HH;

  v8f zero = {};
  v8f acc[NCT];
#pragma unroll
  for (int c = 0; c < NCT; ++c) acc[c] = zero;
  const v16h* Bbase = (const v16h*)Wp + lane + (size_t)ctBase * 32;

  for (int kt = 0; kt < 9; ++kt) {  // full tiles: k0+31 <= 295+24 < 300
    int k0 = kt * 32 + kOff;
    v16h a;
#pragma unroll
    for (int g = 0; g < 2; ++g) {
      float4 m0 = *(const float4*)(maRow + k0 + g * 16);
      float4 m1 = *(const float4*)(maRow + k0 + g * 16 + 4);
      float4 b0 = *(const float4*)(mbRow + k0 + g * 16);
      float4 b1 = *(const float4*)(mbRow + k0 + g * 16 + 4);
      a[g * 8 + 0] = (_Float16)(adjv * m0.x - b0.x);
      a[g * 8 + 1] = (_Float16)(adjv * m0.y - b0.y);
      a[g * 8 + 2] = (_Float16)(adjv * m0.z - b0.z);
      a[g * 8 + 3] = (_Float16)(adjv * m0.w - b0.w);
      a[g * 8 + 4] = (_Float16)(adjv * m1.x - b1.x);
      a[g * 8 + 5] = (_Float16)(adjv * m1.y - b1.y);
      a[g * 8 + 6] = (_Float16)(adjv * m1.z - b1.z);
      a[g * 8 + 7] = (_Float16)(adjv * m1.w - b1.w);
    }
    const v16h* B = Bbase + (size_t)kt * (CTILES * 32);
    v16h bf[NCT];
#pragma unroll
    for (int c = 0; c < NCT; ++c) bf[c] = B[c * 32];
#pragma unroll
    for (int c = 0; c < NCT; ++c) acc[c] = wmma_f16(a, bf[c], acc[c]);
  }
  {  // tail tile kt=9: k in [288, 320), valid < 300
    int k0 = 288 + kOff;
    v16h a;
#pragma unroll
    for (int g = 0; g < 2; ++g)
#pragma unroll
      for (int t = 0; t < 8; ++t) {
        int k = k0 + g * 16 + t;
        float x = 0.f;
        if (k < HH) x = adjv * maRow[k] - mbRow[k];
        a[g * 8 + t] = (_Float16)x;
      }
    const v16h* B = Bbase + (size_t)9 * (CTILES * 32);
    v16h bf[NCT];
#pragma unroll
    for (int c = 0; c < NCT; ++c) bf[c] = B[c * 32];
#pragma unroll
    for (int c = 0; c < NCT; ++c) acc[c] = wmma_f16(a, bf[c], acc[c]);
  }

  int vrow = (lane >> 4) * 8;
  float smv[8];
#pragma unroll
  for (int v = 0; v < 8; ++v)
    smv[v] = sm[(bs * 64 + i) * 64 + (j0 + vrow + v)];
  int rbase = rt * 16 + vrow;
#pragma unroll
  for (int c = 0; c < NCT; ++c) {
    int col = (ctBase + c) * 16 + (lane & 15);
    if (col >= HH) continue;
#pragma unroll
    for (int v = 0; v < 8; ++v) {
      size_t o = (size_t)(rbase + v) * HH + col;
      msgOut[o] = relu_f(acc[c][v] + inputBond[o]) * smv[v];
    }
  }
}

// ---------------------------------------------------------------------------
// agg[bs,m,h] = (sum_n msg[bs,n,m,h]) * sigmoid(max_n msg[bs,n,m,h])
// mode 0: dst += agg (message_atom update)   mode 1: dst = agg
// ---------------------------------------------------------------------------
__global__ void k_agg(const float* __restrict__ msg, float* __restrict__ dst,
                      int mode) {
  int idx = blockIdx.x * blockDim.x + threadIdx.x;
  if (idx >= 32 * 64 * HH) return;
  int h = idx % HH;
  int t = idx / HH;  // bs*64 + m
  const float* base =
      msg + (size_t)(t >> 6) * (64 * 64 * HH) + (size_t)(t & 63) * HH + h;
  float s = 0.f, mx = -3.4e38f;
  for (int n = 0; n < 64; ++n) {
    float v = base[(size_t)n * (64 * HH)];
    s += v;
    mx = fmaxf(mx, v);
  }
  float g = s * sigmoid_f(mx);
  if (mode == 0) dst[idx] += g;
  else dst[idx] = g;
}

// res[bs,n,m] = dot(ma[bs,n,:], ma[bs,m,:]) * adj[bs,n,m]
__global__ void k_res(const float* __restrict__ ma, const float* __restrict__ adj,
                      float* __restrict__ res) {
  int idx = blockIdx.x * blockDim.x + threadIdx.x;
  if (idx >= 32 * 4096) return;
  int m = idx & 63;
  int n = (idx >> 6) & 63;
  int bs = idx >> 12;
  const float4* x = (const float4*)(ma + (size_t)(bs * 64 + n) * HH);
  const float4* y = (const float4*)(ma + (size_t)(bs * 64 + m) * HH);
  float d = 0.f;
  for (int k = 0; k < HH / 4; ++k) {
    float4 a = x[k], b = y[k];
    d += a.x * b.x + a.y * b.y + a.z * b.z + a.w * b.w;
  }
  res[idx] = d * adj[idx];
}

// column softmax over n (axis 2): one thread per (bs,m)
__global__ void k_softmax(const float* __restrict__ res, float* __restrict__ sm) {
  int idx = blockIdx.x * blockDim.x + threadIdx.x;
  if (idx >= 2048) return;
  int m = idx & 63;
  int bs = idx >> 6;
  const float* col = res + (size_t)bs * 4096 + m;
  float mx = -3.4e38f;
  for (int n = 0; n < 64; ++n) mx = fmaxf(mx, col[n * 64]);
  float s = 0.f;
  for (int n = 0; n < 64; ++n) s += __expf(col[n * 64] - mx);
  float inv = 1.f / s;
  float* o = sm + (size_t)bs * 4096 + m;
  for (int n = 0; n < 64; ++n) o[n * 64] = __expf(col[n * 64] - mx) * inv;
}

// ---------------------------------------------------------------------------
// Final GEMM: relu([agg | message_atom | input_atom] @ W_o + b_o), K=900
// ---------------------------------------------------------------------------
template <int NCT>
__global__ void k_final(const float* __restrict__ aggF,
                        const float* __restrict__ ma,
                        const float* __restrict__ ia,
                        const _Float16* __restrict__ Wp,
                        const float* __restrict__ bo, float* __restrict__ out,
                        int ctBase) {
  const int KT = 29, K = 900;  // ceil(900/32)
  int rt = blockIdx.x;
  int lane = threadIdx.x;
  int kOff = (lane >> 4) * 8;
  int r = rt * 16 + (lane & 15);
  v8f zero = {};
  v8f acc[NCT];
#pragma unroll
  for (int c = 0; c < NCT; ++c) acc[c] = zero;
  const v16h* Bbase = (const v16h*)Wp + lane + (size_t)ctBase * 32;

  for (int kt = 0; kt < KT; ++kt) {
    int k0 = kt * 32 + kOff;
    v16h a;
#pragma unroll
    for (int g = 0; g < 2; ++g)
#pragma unroll
      for (int t = 0; t < 8; ++t) {
        int k = k0 + g * 16 + t;
        float x = 0.f;
        if (k < 300) x = aggF[(size_t)r * HH + k];
        else if (k < 600) x = ma[(size_t)r * HH + (k - 300)];
        else if (k < K) x = ia[(size_t)r * HH + (k - 600)];
        a[g * 8 + t] = (_Float16)x;
      }
    const v16h* B = Bbase + (size_t)kt * (CTILES * 32);
    v16h bf[NCT];
#pragma unroll
    for (int c = 0; c < NCT; ++c) bf[c] = B[c * 32];
#pragma unroll
    for (int c = 0; c < NCT; ++c) acc[c] = wmma_f16(a, bf[c], acc[c]);
  }
  int vrow = (lane >> 4) * 8;
#pragma unroll
  for (int c = 0; c < NCT; ++c) {
    int col = (ctBase + c) * 16 + (lane & 15);
    if (col >= HH) continue;
#pragma unroll
    for (int v = 0; v < 8; ++v) {
      int rr = rt * 16 + vrow + v;
      out[(size_t)rr * HH + col] = relu_f(acc[c][v] + bo[col]);
    }
  }
}

// ---------------------------------------------------------------------------
extern "C" void kernel_launch(void* const* d_in, const int* in_sizes, int n_in,
                              void* d_out, int out_size, void* d_ws,
                              size_t ws_size, hipStream_t stream) {
  (void)in_sizes; (void)n_in; (void)out_size; (void)ws_size;
  const float* f_atoms = (const float*)d_in[0];  // [32,64,133]
  const float* f_bonds = (const float*)d_in[1];  // [32,64,64,147]
  const float* adj     = (const float*)d_in[2];  // [32,64,64]
  const float* WiA = (const float*)d_in[3];      // [133,300]
  const float* WiB = (const float*)d_in[4];      // [147,300]
  const float* Wh0 = (const float*)d_in[5];      // [300,300]
  const float* Wh1 = (const float*)d_in[6];      // [300,300]
  const float* Wo  = (const float*)d_in[7];      // [900,300]
  const float* bo  = (const float*)d_in[8];      // [300]
  float* out = (float*)d_out;                    // 614400 + 39321600 floats

  // workspace carve-up (256B aligned)
  char* base = (char*)d_ws;
  size_t off = 0;
  auto alloc = [&](size_t bytes) -> char* {
    char* p = base + off;
    off += (bytes + 255) & ~(size_t)255;
    return p;
  };
  _Float16* pWiA = (_Float16*)alloc((size_t)5 * CTILES * 512 * 2);
  _Float16* pWiB = (_Float16*)alloc((size_t)5 * CTILES * 512 * 2);
  _Float16* pWh0 = (_Float16*)alloc((size_t)10 * CTILES * 512 * 2);
  _Float16* pWh1 = (_Float16*)alloc((size_t)10 * CTILES * 512 * 2);
  _Float16* pWo  = (_Float16*)alloc((size_t)29 * CTILES * 512 * 2);
  float* input_atom   = (float*)alloc(614400ull * 4);
  float* message_atom = (float*)alloc(614400ull * 4);
  float* aggF         = (float*)alloc(614400ull * 4);
  float* res          = (float*)alloc(131072ull * 4);
  float* smb          = (float*)alloc(131072ull * 4);
  float* input_bond   = (float*)alloc(39321600ull * 4);
  float* msgWs        = (float*)alloc(39321600ull * 4);

  // weight packing (tiny)
  k_pack_w<<<(5 * CTILES * 32 + 255) / 256, 256, 0, stream>>>(WiA, pWiA, 133, 5);
  k_pack_w<<<(5 * CTILES * 32 + 255) / 256, 256, 0, stream>>>(WiB, pWiB, 147, 5);
  k_pack_w<<<(10 * CTILES * 32 + 255) / 256, 256, 0, stream>>>(Wh0, pWh0, 300, 10);
  k_pack_w<<<(10 * CTILES * 32 + 255) / 256, 256, 0, stream>>>(Wh1, pWh1, 300, 10);
  k_pack_w<<<(29 * CTILES * 32 + 255) / 256, 256, 0, stream>>>(Wo, pWo, 900, 29);

  // input embeddings: col tile groups 7+6+6 = 19, branch-free inner loops
  auto launch_in = [&](const float* A, const float* sc, const _Float16* Wp,
                       float* o, int rows, int K, int lda, int KTf, int KT) {
    k_gemm_in<7><<<rows / 16, 32, 0, stream>>>(A, sc, Wp, o, 0, K, lda, KTf, KT);
    k_gemm_in<6><<<rows / 16, 32, 0, stream>>>(A, sc, Wp, o, 7, K, lda, KTf, KT);
    k_gemm_in<6><<<rows / 16, 32, 0, stream>>>(A, sc, Wp, o, 13, K, lda, KTf, KT);
  };
  launch_in(f_atoms, nullptr, pWiA, input_atom, 2048, 133, 133, 4, 5);
  launch_in(f_bonds, adj, pWiB, input_bond, 131072, 147, 147, 4, 5);
  hipMemcpyAsync(message_atom, input_atom, 614400ull * 4,
                 hipMemcpyDeviceToDevice, stream);

  // depth loop (message_bond^0 aliases input_bond; final depth writes d_out)
  const float* msgCur = input_bond;
  float* outMB = out + 614400;
  for (int d = 0; d < 2; ++d) {
    k_agg<<<(614400 + 255) / 256, 256, 0, stream>>>(msgCur, message_atom, 0);
    k_res<<<(131072 + 255) / 256, 256, 0, stream>>>(message_atom, adj, res);
    k_softmax<<<(2048 + 255) / 256, 256, 0, stream>>>(res, smb);
    float* dst = (d == 0) ? msgWs : outMB;
    const _Float16* Wh = (d == 0) ? pWh0 : pWh1;
    k_edge_gemm<7><<<8192, 32, 0, stream>>>(msgCur, message_atom, adj,
                                            input_bond, smb, Wh, dst, 0);
    k_edge_gemm<6><<<8192, 32, 0, stream>>>(msgCur, message_atom, adj,
                                            input_bond, smb, Wh, dst, 7);
    k_edge_gemm<6><<<8192, 32, 0, stream>>>(msgCur, message_atom, adj,
                                            input_bond, smb, Wh, dst, 13);
    msgCur = dst;
  }

  // readout
  k_agg<<<(614400 + 255) / 256, 256, 0, stream>>>(msgCur, aggF, 1);
  k_final<7><<<128, 32, 0, stream>>>(aggF, message_atom, input_atom, pWo, bo, out, 0);
  k_final<6><<<128, 32, 0, stream>>>(aggF, message_atom, input_atom, pWo, bo, out, 7);
  k_final<6><<<128, 32, 0, stream>>>(aggF, message_atom, input_atom, pWo, bo, out, 13);
}
